// AttentionBlock_51299089383712
// MI455X (gfx1250) — matrix operand order repaired and checked by reading, not compiled, not run on hardware
//
#include <hip/hip_runtime.h>
#include <hip/hip_bf16.h>

typedef _Float16 half8  __attribute__((ext_vector_type(8)));
typedef _Float16 half16 __attribute__((ext_vector_type(16)));
typedef float    float8 __attribute__((ext_vector_type(8)));
typedef int      i32x4  __attribute__((ext_vector_type(4)));

#define B_  16
#define C_  512
#define T_  1024
#define NH_ 8
#define CH_ 64
#define BH_ 128   // B_*NH_
#define KC_ 512   // GEMM K (= C)

// ---- CDNA5 async memory->LDS path (gfx1250), with portable fallback --------
#if defined(__HIP_DEVICE_COMPILE__) && defined(__gfx1250__) && \
    __has_builtin(__builtin_amdgcn_global_load_async_to_lds_b128) && \
    __has_builtin(__builtin_amdgcn_s_wait_asynccnt)
#define USE_ASYNC 1
#else
#define USE_ASYNC 0
#endif

__device__ __forceinline__ void cp16(const _Float16* g, _Float16* l) {
#if USE_ASYNC
  // generic->AS(1): flat global address IS the global address.
  // generic->AS(3): flat LDS address truncates to the 32-bit LDS offset
  // (ISA 10.2 aperture rules), which is exactly clang's addrspacecast.
  __builtin_amdgcn_global_load_async_to_lds_b128(
      (__attribute__((address_space(1))) i32x4*)(uintptr_t)g,
      (__attribute__((address_space(3))) i32x4*)(unsigned)(uintptr_t)l, 0, 0);
#else
  *(half8*)l = *(const half8*)g;
#endif
}

#if USE_ASYNC
#define WAIT_ASYNC(N) __builtin_amdgcn_s_wait_asynccnt(N)
#else
#define WAIT_ASYNC(N)
#endif

__device__ __forceinline__ half16 cat8(half8 lo, half8 hi) {
  return __builtin_shufflevector(lo, hi, 0,1,2,3,4,5,6,7,8,9,10,11,12,13,14,15);
}

__device__ __forceinline__ float8 wmma_f16(half16 a, half16 b, float8 c) {
  return __builtin_amdgcn_wmma_f32_16x16x32_f16(false, a, false, b, (short)0, c, false, false);
}

// ---------------------------------------------------------------------------
// Kernel 0: one-shot f32->f16 weight conversion (qkv_w, proj_w)
// ---------------------------------------------------------------------------
__global__ __launch_bounds__(256) void cvt_w_kernel(const float* __restrict__ wq,
                                                    const float* __restrict__ wp,
                                                    _Float16* __restrict__ wqh,
                                                    _Float16* __restrict__ wph) {
  const int NQ = 3 * C_ * KC_;                 // 786432
  int e = (blockIdx.x * 256 + threadIdx.x) * 4;
  if (e < NQ) {
    float4 v = *(const float4*)(wq + e);
    wqh[e + 0] = (_Float16)v.x; wqh[e + 1] = (_Float16)v.y;
    wqh[e + 2] = (_Float16)v.z; wqh[e + 3] = (_Float16)v.w;
  } else {
    int p = e - NQ;
    float4 v = *(const float4*)(wp + p);
    wph[p + 0] = (_Float16)v.x; wph[p + 1] = (_Float16)v.y;
    wph[p + 2] = (_Float16)v.z; wph[p + 3] = (_Float16)v.w;
  }
}

// ---------------------------------------------------------------------------
// Kernel 1: GroupNorm (fp32 stats) -> ht[b][t][c] (f16, transposed activation)
// ---------------------------------------------------------------------------
__global__ __launch_bounds__(256) void gn_kernel(const float* __restrict__ x,
                                                 const float* __restrict__ gw,
                                                 const float* __restrict__ gb,
                                                 _Float16* __restrict__ ht) {
  __shared__ float r1[256], r2[256], stats[2];
  const int tid = threadIdx.x;
  const int blk = blockIdx.x;          // b*32 + g
  const int b = blk >> 5, g = blk & 31;
  const float* xb = x + ((size_t)b * C_ + g * 16) * T_;
  float s1 = 0.f, s2 = 0.f;
  for (int idx = tid; idx < 16384; idx += 256) {
    float v = xb[idx];
    s1 += v; s2 += v * v;
  }
  r1[tid] = s1; r2[tid] = s2;
  __syncthreads();
  for (int d = 128; d > 0; d >>= 1) {
    if (tid < d) { r1[tid] += r1[tid + d]; r2[tid] += r2[tid + d]; }
    __syncthreads();
  }
  if (tid == 0) {
    float mean = r1[0] * (1.f / 16384.f);
    float var  = r2[0] * (1.f / 16384.f) - mean * mean;
    stats[0] = mean;
    stats[1] = rsqrtf(var + 1e-5f);
  }
  __syncthreads();
  const float mean = stats[0], rstd = stats[1];
  for (int idx = tid; idx < 16384; idx += 256) {
    int cc = idx & 15, t = idx >> 4;
    int c = g * 16 + cc;
    float y = (xb[cc * T_ + t] - mean) * rstd * gw[c] + gb[c];
    ht[((size_t)b * T_ + t) * C_ + c] = (_Float16)y;
  }
}

// ---------------------------------------------------------------------------
// Shared GEMM staging: A tile 64x64 f16, B^T tile 128x64 f16 (k-stride KC_)
// per-wave async instruction count per batch: 2 (A) + 4 (B) = 6
// ---------------------------------------------------------------------------
__device__ __forceinline__ void stage_tiles(const _Float16* __restrict__ Arow,
                                            const _Float16* __restrict__ Btrow,
                                            _Float16* la, _Float16* lb,
                                            int k0, int tid) {
#pragma unroll
  for (int i = 0; i < 2; ++i) {            // 512 chunks of 16B
    int chunk = tid + i * 256;
    int r = chunk >> 3, c8 = (chunk & 7) * 8;
    cp16(Arow + (size_t)r * KC_ + k0 + c8, la + r * 64 + c8);
  }
#pragma unroll
  for (int i = 0; i < 4; ++i) {            // 1024 chunks of 16B
    int chunk = tid + i * 256;
    int r = chunk >> 3, c8 = (chunk & 7) * 8;
    cp16(Btrow + (size_t)r * KC_ + k0 + c8, lb + r * 64 + c8);
  }
}

// ---------------------------------------------------------------------------
// Kernel 2: QKV GEMM (f16 weights), split + scale into q/k/v layouts
// ---------------------------------------------------------------------------
__global__ __launch_bounds__(256) void qkv_gemm_kernel(const _Float16* __restrict__ Wh,
                                                       const float* __restrict__ bias,
                                                       const _Float16* __restrict__ ht,
                                                       _Float16* __restrict__ q,
                                                       _Float16* __restrict__ k,
                                                       _Float16* __restrict__ v) {
  __shared__ _Float16 lA[2][64 * 64];
  __shared__ _Float16 lB[2][128 * 64];
  const int b    = blockIdx.z;
  const int mblk = blockIdx.y;         // 0..23
  const int nblk = blockIdx.x;         // 0..7
  const int tid  = threadIdx.x;
  const int wave = tid >> 5, lane = tid & 31;
  const int wm = wave & 3, wn = wave >> 2;
  const int lrow = lane & 15;
  const int ksel = (lane < 16) ? 0 : 8;
  const int bsel = (lane < 16) ? 0 : 16;

  float8 acc[4] = {};
  const _Float16* Arow = Wh + (size_t)(mblk * 64) * KC_;
  const _Float16* Bt   = ht + ((size_t)b * T_ + nblk * 128) * KC_;

  const int NK = KC_ / 64;             // 8
  stage_tiles(Arow, Bt, lA[0], lB[0], 0, tid);
  for (int it = 0; it < NK; ++it) {
    const int cur = it & 1;
    if (it + 1 < NK) {
      stage_tiles(Arow, Bt, lA[cur ^ 1], lB[cur ^ 1], (it + 1) * 64, tid);
      WAIT_ASYNC(6);
    } else {
      WAIT_ASYNC(0);
    }
    __syncthreads();
    const _Float16* la = lA[cur];
    const _Float16* lb = lB[cur];
#pragma unroll
    for (int ks = 0; ks < 2; ++ks) {
      const _Float16* ar = la + (wm * 16 + lrow) * 64 + ks * 32;
      const half16 af = cat8(*(const half8*)(ar + bsel), *(const half8*)(ar + bsel + 8));
      half16 bf[4];
#pragma unroll
      for (int j = 0; j < 4; ++j)
        bf[j] = *(const half16*)(lb + (wn * 64 + j * 16 + lrow) * 64 + ks * 32 + bsel);
#pragma unroll
      for (int j = 0; j < 4; ++j) acc[j] = wmma_f16(af, bf[j], acc[j]);
    }
    __syncthreads();
  }

  // epilogue: D element i of this lane is row (ksel+i), col = j*16 + lrow
  const int orow0 = mblk * 64 + wm * 16 + ksel;
  float bias8[8];
#pragma unroll
  for (int i = 0; i < 8; ++i) bias8[i] = bias[orow0 + i];
  const int seg  = orow0 / C_;          // 0=q 1=k 2=v (wave-uniform)
  const int orem = orow0 % C_;
  const int head = orem / CH_, c0 = orem % CH_;
  const int bh = b * NH_ + head;
#pragma unroll
  for (int j = 0; j < 4; ++j) {
    const int t = nblk * 128 + wn * 64 + j * 16 + lrow;
    if (seg == 0) {
      half8 o8;
#pragma unroll
      for (int i = 0; i < 8; ++i) o8[i] = (_Float16)((acc[j][i] + bias8[i]) * 0.125f);
      *(half8*)(q + ((size_t)bh * T_ + t) * CH_ + c0) = o8;
    } else if (seg == 1) {
      half8 o8;
#pragma unroll
      for (int i = 0; i < 8; ++i) o8[i] = (_Float16)(acc[j][i] + bias8[i]);
      *(half8*)(k + ((size_t)bh * T_ + t) * CH_ + c0) = o8;
    } else {
#pragma unroll
      for (int i = 0; i < 8; ++i)
        v[((size_t)bh * CH_ + c0 + i) * T_ + t] = (_Float16)(acc[j][i] + bias8[i]);
    }
  }
}

// ---------------------------------------------------------------------------
// Kernel 3: flash attention per (bh, 64 query rows); 4 waves x 16 rows
// K/V tiles double-buffered; per-wave async instrs per batch: 4 + 4 = 8
// ---------------------------------------------------------------------------
__global__ __launch_bounds__(128) void attn_kernel(const _Float16* __restrict__ q,
                                                   const _Float16* __restrict__ kk,
                                                   const _Float16* __restrict__ vv,
                                                   _Float16* __restrict__ at) {
  __shared__ _Float16 lK[2][64 * 64];    // K tile  [s][c]
  __shared__ _Float16 lV[2][64 * 64];    // V tile  [c][s]
  __shared__ _Float16 lP[4][16 * 64];    // per-wave P transpose buffer [m][s]
  const int bh = blockIdx.y;
  const int mt = blockIdx.x;
  const int tid = threadIdx.x, wave = tid >> 5, lane = tid & 31;
  const int lrow = lane & 15;
  const int ksel = (lane < 16) ? 0 : 8;
  const int bsel = (lane < 16) ? 0 : 16;

  const _Float16* Kbase = kk + (size_t)bh * T_ * CH_;
  const _Float16* Vbase = vv + (size_t)bh * CH_ * T_;

  // load Q A-fragments once (ch=64 -> 2 K-steps of 32)
  const _Float16* qrow = q + ((size_t)bh * T_ + mt * 64 + wave * 16 + lrow) * CH_;
  half16 aq[2];
#pragma unroll
  for (int ks = 0; ks < 2; ++ks)
    aq[ks] = cat8(*(const half8*)(qrow + ks * 32 + bsel),
                  *(const half8*)(qrow + ks * 32 + bsel + 8));

  float8 mi, li = {}, acc[4] = {};
#pragma unroll
  for (int i = 0; i < 8; ++i) mi[i] = -1e30f;
  _Float16* Pw = lP[wave];

  // cooperative K/V staging: 512 + 512 16B chunks, 128 threads
  auto stage_kv = [&](int buf, int s0) {
#pragma unroll
    for (int i = 0; i < 4; ++i) {
      int chunk = tid + i * 128;
      int r = chunk >> 3, o8 = (chunk & 7) * 8;
      cp16(Kbase + (size_t)(s0 + r) * CH_ + o8, lK[buf] + r * 64 + o8);
      cp16(Vbase + (size_t)r * T_ + s0 + o8, lV[buf] + r * 64 + o8);
    }
  };

  const int NS = T_ / 64;              // 16
  stage_kv(0, 0);
  for (int st = 0; st < NS; ++st) {
    const int cur = st & 1;
    if (st + 1 < NS) {
      stage_kv(cur ^ 1, (st + 1) * 64);
      WAIT_ASYNC(8);
    } else {
      WAIT_ASYNC(0);
    }
    __syncthreads();

    // scores: 16 x 64 per wave
    float8 sc[4] = {};
#pragma unroll
    for (int ks = 0; ks < 2; ++ks) {
      half16 bf[4];
#pragma unroll
      for (int j = 0; j < 4; ++j)
        bf[j] = *(const half16*)(lK[cur] + (j * 16 + lrow) * 64 + ks * 32 + bsel);
#pragma unroll
      for (int j = 0; j < 4; ++j) sc[j] = wmma_f16(aq[ks], bf[j], sc[j]);
    }

    // online softmax (rows live in 16-lane halves; element i <-> row ksel+i)
    float8 mt8;
#pragma unroll
    for (int i = 0; i < 8; ++i)
      mt8[i] = fmaxf(fmaxf(sc[0][i], sc[1][i]), fmaxf(sc[2][i], sc[3][i]));
#pragma unroll
    for (int d = 1; d < 16; d <<= 1)
#pragma unroll
      for (int i = 0; i < 8; ++i) mt8[i] = fmaxf(mt8[i], __shfl_xor(mt8[i], d, 16));
    float8 mnew, corr, lt = {};
#pragma unroll
    for (int i = 0; i < 8; ++i) { mnew[i] = fmaxf(mi[i], mt8[i]); corr[i] = __expf(mi[i] - mnew[i]); }
#pragma unroll
    for (int j = 0; j < 4; ++j)
#pragma unroll
      for (int i = 0; i < 8; ++i) { float p = __expf(sc[j][i] - mnew[i]); sc[j][i] = p; lt[i] += p; }
#pragma unroll
    for (int d = 1; d < 16; d <<= 1)
#pragma unroll
      for (int i = 0; i < 8; ++i) lt[i] += __shfl_xor(lt[i], d, 16);
#pragma unroll
    for (int i = 0; i < 8; ++i) { li[i] = li[i] * corr[i] + lt[i]; mi[i] = mnew[i]; }
#pragma unroll
    for (int j = 0; j < 4; ++j)
#pragma unroll
      for (int i = 0; i < 8; ++i) acc[j][i] *= corr[i];

    // transpose P through per-wave LDS region: [m=row][s]
#pragma unroll
    for (int j = 0; j < 4; ++j)
#pragma unroll
      for (int i = 0; i < 8; ++i)
        Pw[(ksel + i) * 64 + j * 16 + lrow] = (_Float16)sc[j][i];

    // acc += P * V   (K-dim = s, 2 steps of 32)
#pragma unroll
    for (int ks = 0; ks < 2; ++ks) {
      const _Float16* pr = Pw + lrow * 64 + ks * 32;
      const half16 ap = cat8(*(const half8*)(pr + bsel), *(const half8*)(pr + bsel + 8));
      half16 bf[4];
#pragma unroll
      for (int jc = 0; jc < 4; ++jc)
        bf[jc] = *(const half16*)(lV[cur] + (jc * 16 + lrow) * 64 + ks * 32 + bsel);
#pragma unroll
      for (int jc = 0; jc < 4; ++jc) acc[jc] = wmma_f16(ap, bf[jc], acc[jc]);
    }
    __syncthreads();
  }

  // finalize: at[b][t][head*64 + c] f16
  float8 inv;
#pragma unroll
  for (int i = 0; i < 8; ++i) inv[i] = 1.f / li[i];
  const int b = bh >> 3, head = bh & 7;
#pragma unroll
  for (int jc = 0; jc < 4; ++jc)
#pragma unroll
    for (int i = 0; i < 8; ++i) {
      int tg = mt * 64 + wave * 16 + ksel + i;
      int cg = head * CH_ + jc * 16 + lrow;
      at[((size_t)b * T_ + tg) * C_ + cg] = (_Float16)(acc[jc][i] * inv[i]);
    }
}

// ---------------------------------------------------------------------------
// Kernel 4: proj GEMM (f16 weights) + bias + residual -> fp32 out[b][o][t]
// ---------------------------------------------------------------------------
__global__ __launch_bounds__(256) void proj_gemm_kernel(const _Float16* __restrict__ Wh,
                                                        const float* __restrict__ bias,
                                                        const _Float16* __restrict__ at,
                                                        const float* __restrict__ x,
                                                        float* __restrict__ out) {
  __shared__ _Float16 lA[2][64 * 64];
  __shared__ _Float16 lB[2][128 * 64];
  const int b    = blockIdx.z;
  const int mblk = blockIdx.y;         // 0..7
  const int nblk = blockIdx.x;         // 0..7
  const int tid  = threadIdx.x;
  const int wave = tid >> 5, lane = tid & 31;
  const int wm = wave & 3, wn = wave >> 2;
  const int lrow = lane & 15;
  const int ksel = (lane < 16) ? 0 : 8;
  const int bsel = (lane < 16) ? 0 : 16;

  float8 acc[4] = {};
  const _Float16* Arow = Wh + (size_t)(mblk * 64) * KC_;
  const _Float16* Bt   = at + ((size_t)b * T_ + nblk * 128) * KC_;

  const int NK = KC_ / 64;
  stage_tiles(Arow, Bt, lA[0], lB[0], 0, tid);
  for (int it = 0; it < NK; ++it) {
    const int cur = it & 1;
    if (it + 1 < NK) {
      stage_tiles(Arow, Bt, lA[cur ^ 1], lB[cur ^ 1], (it + 1) * 64, tid);
      WAIT_ASYNC(6);
    } else {
      WAIT_ASYNC(0);
    }
    __syncthreads();
    const _Float16* la = lA[cur];
    const _Float16* lb = lB[cur];
#pragma unroll
    for (int ks = 0; ks < 2; ++ks) {
      const _Float16* ar = la + (wm * 16 + lrow) * 64 + ks * 32;
      const half16 af = cat8(*(const half8*)(ar + bsel), *(const half8*)(ar + bsel + 8));
      half16 bf[4];
#pragma unroll
      for (int j = 0; j < 4; ++j)
        bf[j] = *(const half16*)(lb + (wn * 64 + j * 16 + lrow) * 64 + ks * 32 + bsel);
#pragma unroll
      for (int j = 0; j < 4; ++j) acc[j] = wmma_f16(af, bf[j], acc[j]);
    }
    __syncthreads();
  }

  const int orow0 = mblk * 64 + wm * 16 + ksel;
#pragma unroll
  for (int j = 0; j < 4; ++j) {
    const int t = nblk * 128 + wn * 64 + j * 16 + lrow;
#pragma unroll
    for (int i = 0; i < 8; ++i) {
      const int o = orow0 + i;
      const size_t idx = ((size_t)b * C_ + o) * T_ + t;
      out[idx] = x[idx] + bias[o] + acc[j][i];
    }
  }
}

// ---------------------------------------------------------------------------
extern "C" void kernel_launch(void* const* d_in, const int* in_sizes, int n_in,
                              void* d_out, int out_size, void* d_ws, size_t ws_size,
                              hipStream_t stream) {
  const float* x      = (const float*)d_in[0];
  const float* norm_w = (const float*)d_in[1];
  const float* norm_b = (const float*)d_in[2];
  const float* qkv_w  = (const float*)d_in[3];
  const float* qkv_b  = (const float*)d_in[4];
  const float* proj_w = (const float*)d_in[5];
  const float* proj_b = (const float*)d_in[6];
  float* out = (float*)d_out;

  char* ws = (char*)d_ws;
  const size_t SZ = (size_t)B_ * C_ * T_ * sizeof(_Float16);  // 16 MB each
  _Float16* ht  = (_Float16*)(ws);
  _Float16* q   = (_Float16*)(ws + SZ);
  _Float16* k   = (_Float16*)(ws + 2 * SZ);
  _Float16* v   = (_Float16*)(ws + 3 * SZ);
  _Float16* at  = (_Float16*)(ws + 4 * SZ);
  _Float16* wqh = (_Float16*)(ws + 5 * SZ);                       // 1.5 MB
  _Float16* wph = (_Float16*)(ws + 5 * SZ + (size_t)2 * 1024 * 1024);

  cvt_w_kernel<<<dim3((3 * C_ * KC_ + C_ * KC_) / (256 * 4)), 256, 0, stream>>>(
      qkv_w, proj_w, wqh, wph);
  gn_kernel<<<dim3(B_ * 32), 256, 0, stream>>>(x, norm_w, norm_b, ht);
  qkv_gemm_kernel<<<dim3(T_ / 128, (3 * C_) / 64, B_), 256, 0, stream>>>(wqh, qkv_b, ht, q, k, v);
  attn_kernel<<<dim3(T_ / 64, BH_), 128, 0, stream>>>(q, k, v, at);
  proj_gemm_kernel<<<dim3(T_ / 128, C_ / 64, B_), 256, 0, stream>>>(wph, proj_b, at, x, out);
}
